// FlowMatchingModel_420906795770
// MI455X (gfx1250) — matrix operand order repaired and checked by reading, not compiled, not run on hardware
//
#include <hip/hip_runtime.h>

#define NN 10000
#define NE 320000
#define HH 128
#define NL 4
#define FM_EPS 1e-08f

typedef __attribute__((ext_vector_type(16))) _Float16 v16h;
typedef __attribute__((ext_vector_type(8)))  _Float16 v8h;
typedef __attribute__((ext_vector_type(8)))  float    v8f;

__device__ __forceinline__ float fm_silu(float v) { return v / (1.0f + __expf(-v)); }

// A-fragment (16x32 f16): lane L covers row M=L&15; half=L>>4.
// elem e (0..7) -> K = K0 + half*8 + e ; elem e (8..15) -> K = K0 + 16 + half*8 + (e-8)
__device__ __forceinline__ v16h fm_load_a(const _Float16* p /*row base + K0*/, int half) {
    v8h lo = *(const v8h*)(p + half * 8);
    v8h hi = *(const v8h*)(p + 16 + half * 8);
    v16h a;
#pragma unroll
    for (int j = 0; j < 8; ++j) { a[j] = lo[j]; a[8 + j] = hi[j]; }
    return a;
}

// Same but converting from f32 source
__device__ __forceinline__ v16h fm_load_a_f32(const float* p, int half) {
    const float* p0 = p + half * 8;
    const float* p1 = p + 16 + half * 8;
    v16h a;
#pragma unroll
    for (int j = 0; j < 8; ++j) { a[j] = (_Float16)p0[j]; a[8 + j] = (_Float16)p1[j]; }
    return a;
}

// B-fragment from packed weights: frag = packed[((nTile*numK + kc)*32 + lane)*16 .. +15]
__device__ __forceinline__ v16h fm_load_b(const _Float16* packed, int nTile, int kc, int numK, int lane) {
    const _Float16* q = packed + (((size_t)(nTile * numK + kc) * 32 + lane) << 4);
    v8h lo = *(const v8h*)q;
    v8h hi = *(const v8h*)(q + 8);
    v16h b;
#pragma unroll
    for (int j = 0; j < 8; ++j) { b[j] = lo[j]; b[8 + j] = hi[j]; }
    return b;
}

// ---------- weight packing: src (K x 128 f32, optional skipped row) -> fragment order f16 ----------
__global__ void fm_pack_w(const float* __restrict__ src, _Float16* __restrict__ dst,
                          int numK, int skipAfter) {
    int tid = blockIdx.x * blockDim.x + threadIdx.x;
    int total = numK * 8 * 32 * 16;
    if (tid >= total) return;
    int e = tid & 15;
    int lane = (tid >> 4) & 31;
    int rest = tid >> 9;
    int kc = rest % numK;
    int nTile = rest / numK;
    int K = kc * 32 + ((e >> 3) << 4) + ((lane >> 4) << 3) + (e & 7);        // B layout: K = 16*(e/8) + 8*half + e%8 within chunk
    int srcRow = K + ((skipAfter >= 0 && K >= skipAfter) ? 1 : 0);
    int n = nTile * 16 + (lane & 15);
    dst[tid] = (_Float16)src[(size_t)srcRow * 128 + n];
}

__global__ void fm_extract_row256(const float* __restrict__ src, float* __restrict__ dst) {
    int j = threadIdx.x;
    dst[j] = src[256 * 128 + j];
}

// ---------- prep kernels ----------
__global__ void fm_temb(const float* __restrict__ t, const float* __restrict__ w1,
                        const float* __restrict__ b1, const float* __restrict__ w2,
                        const float* __restrict__ b2, float* __restrict__ temb) {
    __shared__ float hid[128];
    int j = threadIdx.x;
    float hv = t[0] * w1[j] + b1[j];
    hid[j] = fm_silu(hv);
    __syncthreads();
    float acc = b2[j];
    for (int k = 0; k < 128; ++k) acc += hid[k] * w2[k * 128 + j];
    temb[j] = acc;
}

__global__ void fm_hfeat(const float* __restrict__ h, const float* __restrict__ w,
                         const float* __restrict__ b, const float* __restrict__ temb,
                         float* __restrict__ hf32, _Float16* __restrict__ hf16) {
    int tid = blockIdx.x * blockDim.x + threadIdx.x;
    if (tid >= NN * 128) return;
    int i = tid >> 7, j = tid & 127;
    float acc = b[j] + temb[j];
    const float* hr = h + (size_t)i * 64;
    for (int k = 0; k < 64; ++k) acc += hr[k] * w[k * 128 + j];
    hf32[tid] = acc;
    hf16[tid] = (_Float16)acc;
}

__global__ void fm_efeat(const float* __restrict__ ea, const float* __restrict__ w,
                         const float* __restrict__ b, _Float16* __restrict__ ef16) {
    size_t tid = (size_t)blockIdx.x * blockDim.x + threadIdx.x;
    if (tid >= (size_t)NE * 128) return;
    size_t i = tid >> 7; int j = (int)(tid & 127);
    float acc = b[j];
    const float* er = ea + i * 16;
    for (int k = 0; k < 16; ++k) acc += er[k] * w[k * 128 + j];
    ef16[tid] = (_Float16)acc;
}

__global__ void fm_copy(const float* __restrict__ src, float* __restrict__ dst, int n) {
    int tid = blockIdx.x * blockDim.x + threadIdx.x;
    if (tid < n) dst[tid] = src[tid];
}
__global__ void fm_zero(float* __restrict__ p, int n) {
    int tid = blockIdx.x * blockDim.x + threadIdx.x;
    if (tid < n) p[tid] = 0.0f;
}
__global__ void fm_xupdate(float* __restrict__ xcur, const float* __restrict__ aggx,
                           const float* __restrict__ mask) {
    int tid = blockIdx.x * blockDim.x + threadIdx.x;
    if (tid < NN * 3) xcur[tid] += aggx[tid] * mask[tid / 3];
}
__global__ void fm_final(const float* __restrict__ xcur, const float* __restrict__ xin,
                         const float* __restrict__ mask, float* __restrict__ out) {
    int tid = blockIdx.x * blockDim.x + threadIdx.x;
    if (tid < NN * 3) out[tid] = (xcur[tid] - xin[tid]) * mask[tid / 3];
}

// ---------- edge kernel: one wave per 16-edge tile, all GEMMs via WMMA ----------
__global__ void __launch_bounds__(32) fm_edge(
    const int* __restrict__ row, const int* __restrict__ col,
    const float* __restrict__ xcur,
    const _Float16* __restrict__ hf16, const _Float16* __restrict__ ef16,
    const _Float16* __restrict__ w1p, const float* __restrict__ w1d, const float* __restrict__ b1,
    const _Float16* __restrict__ w2p, const float* __restrict__ b2,
    const _Float16* __restrict__ c1p, const float* __restrict__ cb,
    const float* __restrict__ c2,
    float* __restrict__ aggmsg, float* __restrict__ aggx) {
    __shared__ __align__(16) _Float16 stage[16 * 128];
    __shared__ float sdist[16];
    __shared__ float sdiff[16][3];
    __shared__ int   srow[16];
    __shared__ float sw[16];

    const int lane = threadIdx.x;
    const int m = lane & 15, half = lane >> 4;
    const int eIdx = blockIdx.x * 16 + m;
    const int r = row[eIdx];
    const int c = col[eIdx];
    if (half == 0) {
        float dx = xcur[r * 3 + 0] - xcur[c * 3 + 0];
        float dy = xcur[r * 3 + 1] - xcur[c * 3 + 1];
        float dz = xcur[r * 3 + 2] - xcur[c * 3 + 2];
        sdiff[m][0] = dx; sdiff[m][1] = dy; sdiff[m][2] = dz;
        sdist[m] = dx * dx + dy * dy + dz * dz;
        srow[m] = r;
    }
    __syncthreads();

    const v8f vzero = {0.f, 0.f, 0.f, 0.f, 0.f, 0.f, 0.f, 0.f};
    const _Float16* rowH = hf16 + (size_t)r * 128;
    const _Float16* colH = hf16 + (size_t)c * 128;
    const _Float16* edgH = ef16 + (size_t)eIdx * 128;

    // ---- GEMM1: edge_in(16x384) @ w1p -> 16x128 ----
    v8f acc[8];
#pragma unroll
    for (int t = 0; t < 8; ++t) acc[t] = vzero;
    for (int kc = 0; kc < 12; ++kc) {
        const _Float16* src = (kc < 4) ? (rowH + kc * 32)
                           : (kc < 8) ? (colH + (kc - 4) * 32)
                                      : (edgH + (kc - 8) * 32);
        v16h a = fm_load_a(src, half);
#pragma unroll
        for (int t = 0; t < 8; ++t) {
            v16h b = fm_load_b(w1p, t, kc, 12, lane);
            acc[t] = __builtin_amdgcn_wmma_f32_16x16x32_f16(false, a, false, b,
                                                            (short)0, acc[t], false, false);
        }
    }
    // epilogue: + dist_sq*w1_dist + b1, silu -> stage (f16)
    float sd[8];
#pragma unroll
    for (int v = 0; v < 8; ++v) sd[v] = sdist[v + 8 * half];
#pragma unroll
    for (int t = 0; t < 8; ++t) {
        int Nn = t * 16 + m;
        float bb = b1[Nn], wd = w1d[Nn];
#pragma unroll
        for (int v = 0; v < 8; ++v) {
            float val = acc[t][v] + sd[v] * wd + bb;
            stage[(v + 8 * half) * 128 + Nn] = (_Float16)fm_silu(val);
        }
    }
    __syncthreads();

    // ---- GEMM2: (16x128) @ w2p -> m (16x128) ----
#pragma unroll
    for (int t = 0; t < 8; ++t) acc[t] = vzero;
    for (int kc = 0; kc < 4; ++kc) {
        v16h a = fm_load_a(stage + m * 128 + kc * 32, half);
#pragma unroll
        for (int t = 0; t < 8; ++t) {
            v16h b = fm_load_b(w2p, t, kc, 4, lane);
            acc[t] = __builtin_amdgcn_wmma_f32_16x16x32_f16(false, a, false, b,
                                                            (short)0, acc[t], false, false);
        }
    }
    __syncthreads();  // all reads of stage done before overwrite
    int rdest[8];
#pragma unroll
    for (int v = 0; v < 8; ++v) rdest[v] = srow[v + 8 * half];
#pragma unroll
    for (int t = 0; t < 8; ++t) {
        int Nn = t * 16 + m;
        float bb = b2[Nn];
#pragma unroll
        for (int v = 0; v < 8; ++v) {
            float val = fm_silu(acc[t][v] + bb);                 // m message value
            atomicAdd(&aggmsg[(size_t)rdest[v] * 128 + Nn], val); // segment_sum(m)
            stage[(v + 8 * half) * 128 + Nn] = (_Float16)val;
        }
    }
    __syncthreads();

    // ---- GEMM3: m @ c1p -> 16x128, silu, then dot with c2 (128->1), tanh ----
#pragma unroll
    for (int t = 0; t < 8; ++t) acc[t] = vzero;
    for (int kc = 0; kc < 4; ++kc) {
        v16h a = fm_load_a(stage + m * 128 + kc * 32, half);
#pragma unroll
        for (int t = 0; t < 8; ++t) {
            v16h b = fm_load_b(c1p, t, kc, 4, lane);
            acc[t] = __builtin_amdgcn_wmma_f32_16x16x32_f16(false, a, false, b,
                                                            (short)0, acc[t], false, false);
        }
    }
    float part[8];
#pragma unroll
    for (int v = 0; v < 8; ++v) part[v] = 0.0f;
#pragma unroll
    for (int t = 0; t < 8; ++t) {
        int Nn = t * 16 + m;
        float cbv = cb[Nn], c2v = c2[Nn];
#pragma unroll
        for (int v = 0; v < 8; ++v) part[v] += fm_silu(acc[t][v] + cbv) * c2v;
    }
#pragma unroll
    for (int off = 1; off < 16; off <<= 1) {
#pragma unroll
        for (int v = 0; v < 8; ++v) part[v] += __shfl_xor(part[v], off, 32);
    }
    if (m == 0) {
#pragma unroll
        for (int v = 0; v < 8; ++v) sw[v + 8 * half] = tanhf(part[v]);
    }
    __syncthreads();

    if (half == 0) {
        float dist = sqrtf(sdist[m] + FM_EPS);
        float wv = sw[m] / (dist + FM_EPS);
        int rr = srow[m];
        atomicAdd(&aggx[rr * 3 + 0], sdiff[m][0] * wv);
        atomicAdd(&aggx[rr * 3 + 1], sdiff[m][1] * wv);
        atomicAdd(&aggx[rr * 3 + 2], sdiff[m][2] * wv);
    }
}

// ---------- node kernel: h_new = hf + silu([hf, agg_msg] @ n1 + nb1) @ n2 + nb2 ----------
__global__ void __launch_bounds__(32) fm_node(
    const _Float16* __restrict__ n1p, const float* __restrict__ nb1,
    const _Float16* __restrict__ n2p, const float* __restrict__ nb2,
    const float* __restrict__ aggmsg,
    float* __restrict__ hf32, _Float16* __restrict__ hf16) {
    __shared__ __align__(16) _Float16 stage[16 * 128];
    const int lane = threadIdx.x;
    const int m = lane & 15, half = lane >> 4;
    const int node = blockIdx.x * 16 + m;
    const _Float16* hrow = hf16 + (size_t)node * 128;
    const float* arow = aggmsg + (size_t)node * 128;

    const v8f vzero = {0.f, 0.f, 0.f, 0.f, 0.f, 0.f, 0.f, 0.f};
    v8f acc[8];
#pragma unroll
    for (int t = 0; t < 8; ++t) acc[t] = vzero;
    for (int kc = 0; kc < 8; ++kc) {
        v16h a = (kc < 4) ? fm_load_a(hrow + kc * 32, half)
                          : fm_load_a_f32(arow + (kc - 4) * 32, half);
#pragma unroll
        for (int t = 0; t < 8; ++t) {
            v16h b = fm_load_b(n1p, t, kc, 8, lane);
            acc[t] = __builtin_amdgcn_wmma_f32_16x16x32_f16(false, a, false, b,
                                                            (short)0, acc[t], false, false);
        }
    }
#pragma unroll
    for (int t = 0; t < 8; ++t) {
        int Nn = t * 16 + m;
        float bb = nb1[Nn];
#pragma unroll
        for (int v = 0; v < 8; ++v)
            stage[(v + 8 * half) * 128 + Nn] = (_Float16)fm_silu(acc[t][v] + bb);
    }
    __syncthreads();

#pragma unroll
    for (int t = 0; t < 8; ++t) acc[t] = vzero;
    for (int kc = 0; kc < 4; ++kc) {
        v16h a = fm_load_a(stage + m * 128 + kc * 32, half);
#pragma unroll
        for (int t = 0; t < 8; ++t) {
            v16h b = fm_load_b(n2p, t, kc, 4, lane);
            acc[t] = __builtin_amdgcn_wmma_f32_16x16x32_f16(false, a, false, b,
                                                            (short)0, acc[t], false, false);
        }
    }
#pragma unroll
    for (int t = 0; t < 8; ++t) {
        int Nn = t * 16 + m;
        float bb = nb2[Nn];
#pragma unroll
        for (int v = 0; v < 8; ++v) {
            int nd = blockIdx.x * 16 + (v + 8 * half);
            size_t o = (size_t)nd * 128 + Nn;
            float val = hf32[o] + acc[t][v] + bb;
            hf32[o] = val;
            hf16[o] = (_Float16)val;
        }
    }
}

extern "C" void kernel_launch(void* const* d_in, const int* in_sizes, int n_in,
                              void* d_out, int out_size, void* d_ws, size_t ws_size,
                              hipStream_t stream) {
    (void)in_sizes; (void)n_in; (void)out_size; (void)ws_size;
    const float* h     = (const float*)d_in[0];
    const float* x     = (const float*)d_in[1];
    const int*   eidx  = (const int*)d_in[2];
    const float* eattr = (const float*)d_in[3];
    const float* t     = (const float*)d_in[4];
    const float* mask  = (const float*)d_in[5];
    const float* tw1   = (const float*)d_in[6];
    const float* tb1   = (const float*)d_in[7];
    const float* tw2   = (const float*)d_in[8];
    const float* tb2   = (const float*)d_in[9];
    const float* newb  = (const float*)d_in[10];
    const float* neb   = (const float*)d_in[11];
    const float* eew   = (const float*)d_in[12];
    const float* eeb   = (const float*)d_in[13];
    const float* Wew1  = (const float*)d_in[14];
    const float* Web1  = (const float*)d_in[15];
    const float* Wew2  = (const float*)d_in[16];
    const float* Web2  = (const float*)d_in[17];
    const float* Wcw1  = (const float*)d_in[18];
    const float* Wcb1  = (const float*)d_in[19];
    const float* Wcw2  = (const float*)d_in[20];
    const float* Wnw1  = (const float*)d_in[21];
    const float* Wnb1  = (const float*)d_in[22];
    const float* Wnw2  = (const float*)d_in[23];
    const float* Wnb2  = (const float*)d_in[24];
    const int* row = eidx;
    const int* col = eidx + NE;

    // workspace carving (256B aligned)
    char* p = (char*)d_ws;
    auto take = [&](size_t bytes) -> void* {
        void* r = (void*)p;
        p += (bytes + 255) & ~(size_t)255;
        return r;
    };
    float*     hf32   = (float*)take((size_t)NN * 128 * 4);
    _Float16*  hf16   = (_Float16*)take((size_t)NN * 128 * 2);
    _Float16*  ef16   = (_Float16*)take((size_t)NE * 128 * 2);
    float*     xcur   = (float*)take((size_t)NN * 3 * 4);
    float*     aggmsg = (float*)take((size_t)NN * 128 * 4);
    float*     aggx   = (float*)take((size_t)NN * 3 * 4);
    float*     temb   = (float*)take(128 * 4);
    const int W1P = 12 * 8 * 512, W2P = 4 * 8 * 512, N1P = 8 * 8 * 512;
    _Float16*  w1p = (_Float16*)take((size_t)NL * W1P * 2);
    float*     w1d = (float*)take((size_t)NL * 128 * 4);
    _Float16*  w2p = (_Float16*)take((size_t)NL * W2P * 2);
    _Float16*  c1p = (_Float16*)take((size_t)NL * W2P * 2);
    _Float16*  n1p = (_Float16*)take((size_t)NL * N1P * 2);
    _Float16*  n2p = (_Float16*)take((size_t)NL * W2P * 2);

    // pack weights into WMMA B-fragment order
    for (int l = 0; l < NL; ++l) {
        fm_pack_w<<<(W1P + 255) / 256, 256, 0, stream>>>(Wew1 + (size_t)l * 385 * 128, w1p + (size_t)l * W1P, 12, 256);
        fm_extract_row256<<<1, 128, 0, stream>>>(Wew1 + (size_t)l * 385 * 128, w1d + l * 128);
        fm_pack_w<<<(W2P + 255) / 256, 256, 0, stream>>>(Wew2 + (size_t)l * 128 * 128, w2p + (size_t)l * W2P, 4, -1);
        fm_pack_w<<<(W2P + 255) / 256, 256, 0, stream>>>(Wcw1 + (size_t)l * 128 * 128, c1p + (size_t)l * W2P, 4, -1);
        fm_pack_w<<<(N1P + 255) / 256, 256, 0, stream>>>(Wnw1 + (size_t)l * 256 * 128, n1p + (size_t)l * N1P, 8, -1);
        fm_pack_w<<<(W2P + 255) / 256, 256, 0, stream>>>(Wnw2 + (size_t)l * 128 * 128, n2p + (size_t)l * W2P, 4, -1);
    }

    // embeddings
    fm_temb<<<1, 128, 0, stream>>>(t, tw1, tb1, tw2, tb2, temb);
    fm_hfeat<<<(NN * 128 + 255) / 256, 256, 0, stream>>>(h, newb, neb, temb, hf32, hf16);
    fm_efeat<<<(int)(((size_t)NE * 128 + 255) / 256), 256, 0, stream>>>(eattr, eew, eeb, ef16);
    fm_copy<<<(NN * 3 + 255) / 256, 256, 0, stream>>>(x, xcur, NN * 3);

    for (int l = 0; l < NL; ++l) {
        fm_zero<<<(NN * 128 + 255) / 256, 256, 0, stream>>>(aggmsg, NN * 128);
        fm_zero<<<(NN * 3 + 255) / 256, 256, 0, stream>>>(aggx, NN * 3);
        fm_edge<<<NE / 16, 32, 0, stream>>>(row, col, xcur, hf16, ef16,
                                            w1p + (size_t)l * W1P, w1d + l * 128, Web1 + l * 128,
                                            w2p + (size_t)l * W2P, Web2 + l * 128,
                                            c1p + (size_t)l * W2P, Wcb1 + l * 128,
                                            Wcw2 + l * 128,
                                            aggmsg, aggx);
        fm_node<<<NN / 16, 32, 0, stream>>>(n1p + (size_t)l * N1P, Wnb1 + l * 128,
                                            n2p + (size_t)l * W2P, Wnb2 + l * 128,
                                            aggmsg, hf32, hf16);
        fm_xupdate<<<(NN * 3 + 255) / 256, 256, 0, stream>>>(xcur, aggx, mask);
    }
    fm_final<<<(NN * 3 + 255) / 256, 256, 0, stream>>>(xcur, x, mask, (float*)d_out);
}
